// Attention_22067541967803
// MI455X (gfx1250) — matrix operand order repaired and checked by reading, not compiled, not run on hardware
//
#include <hip/hip_runtime.h>

// ---------------------------------------------------------------------------
// Types for CDNA5 WMMA (wave32): 16x16x32 f16 -> f32
// ---------------------------------------------------------------------------
typedef __attribute__((ext_vector_type(16))) _Float16 v16h;
typedef __attribute__((ext_vector_type(8)))  _Float16 v8h;
typedef __attribute__((ext_vector_type(2)))  _Float16 v2h;
typedef __attribute__((ext_vector_type(8)))  float    v8f;
typedef __attribute__((ext_vector_type(4)))  float    v4f;

static __device__ __forceinline__ v16h ld_frag(const _Float16* p, int hi_off_halfs) {
  union { v16h v; v8h h[2]; } u;
  u.h[0] = *(const v8h*)(p);
  u.h[1] = *(const v8h*)(p + hi_off_halfs);
  return u.v;
}

static __device__ __forceinline__ v8f wmma_f16(v16h a, v16h b, v8f c) {
  return __builtin_amdgcn_wmma_f32_16x16x32_f16(
      false, a, false, b, (short)0, c, false, false);
}

static __device__ __forceinline__ v4f ld4(const float* p) { return *(const v4f*)p; }

// ---------------------------------------------------------------------------
// WMMA GEMM:  C[M,N] = A[M,K] * B[K,N] (+ bias[N])
//   A16P:   A is f16, staged via global_load_async_to_lds_b128 (ASYNCcnt path)
//           else A fp32, register-staged + converted (two b128 DS stores)
//   OUT16:  f16 output, else fp32
//   TRANS:  store output transposed: out16[col*M + row] (packed b128 stores)
// Block 256 thr = 8 waves (2Mx4N), block tile 128x128, wave tile 64x32,
// double-buffered LDS, one barrier per 32-wide K chunk.
// ---------------------------------------------------------------------------
template<bool A16P, bool OUT16, bool TRANS>
__global__ __launch_bounds__(256)
void wmma_gemm_kernel(const float* __restrict__ A32, const _Float16* __restrict__ A16,
                      const float* __restrict__ Bw,  const float* __restrict__ bias,
                      _Float16* __restrict__ out16,  float* __restrict__ out32,
                      int M, int N, int K) {
  __shared__ _Float16 Alds[2][128 * 32];   // [m][k]
  __shared__ _Float16 Blds[2][128 * 32];   // transposed: [n][k]

  const int tid  = threadIdx.x;
  const int lane = tid & 31;
  const int wave = tid >> 5;
  const int lg   = lane >> 4;
  const int ln   = lane & 15;
  const int waveM = (wave >> 2) * 64;
  const int waveN = (wave & 3)  * 32;
  const int blockM = blockIdx.y * 128;
  const int blockN = blockIdx.x * 128;

  v8f acc[4][2];
  #pragma unroll
  for (int i = 0; i < 4; i++)
    #pragma unroll
    for (int j = 0; j < 2; j++) acc[i][j] = v8f{};

  const int am  = tid & 127;            // A row
  const int aks = (tid >> 7) * 16;      // A 16-half segment
  const int bkp = (tid & 15) * 2;       // B k-pair
  const int bn0 = (tid >> 4) * 8;       // B 8 n values

  const uint64_t a16base = (uint64_t)(uintptr_t)A16;

  float ar[16];
  v4f   br[4];

  auto a_async = [&](int kb, int buf) {
    uint32_t lds = (uint32_t)(uintptr_t)&Alds[buf][am * 32 + aks];
    uint32_t go  = (uint32_t)(((size_t)(blockM + am) * K + kb + aks) * 2);
    asm volatile("global_load_async_to_lds_b128 %0, %1, %2 offset:0\n\t"
                 "global_load_async_to_lds_b128 %0, %1, %2 offset:16"
                 :: "v"(lds), "v"(go), "s"(a16base) : "memory");
  };
  auto a_load = [&](int kb) {
    const float* src = A32 + (size_t)(blockM + am) * K + kb + aks;
    #pragma unroll
    for (int q = 0; q < 4; q++) { v4f t = ld4(src + q * 4);
      #pragma unroll
      for (int j = 0; j < 4; j++) ar[q * 4 + j] = t[j]; }
  };
  auto a_store = [&](int buf) {
    v8h h0, h1;
    #pragma unroll
    for (int j = 0; j < 8; j++) { h0[j] = (_Float16)ar[j]; h1[j] = (_Float16)ar[8 + j]; }
    *(v8h*)&Alds[buf][am * 32 + aks]     = h0;
    *(v8h*)&Alds[buf][am * 32 + aks + 8] = h1;
  };
  auto b_load = [&](int kb) {
    const float* b0 = Bw + (size_t)(kb + bkp) * N + blockN + bn0;
    br[0] = ld4(b0); br[1] = ld4(b0 + 4);
    br[2] = ld4(b0 + N); br[3] = ld4(b0 + N + 4);
  };
  auto b_store = [&](int buf) {
    #pragma unroll
    for (int j = 0; j < 8; j++) {
      v2h p; p[0] = (_Float16)br[j >> 2][j & 3]; p[1] = (_Float16)br[2 + (j >> 2)][j & 3];
      *(v2h*)&Blds[buf][(bn0 + j) * 32 + bkp] = p;
    }
  };

  // prologue
  if constexpr (A16P) {
    a_async(0, 0);
    b_load(0); b_store(0);
    asm volatile("s_wait_asynccnt 0x0" ::: "memory");
  } else {
    a_load(0); a_store(0);
    b_load(0); b_store(0);
  }
  __syncthreads();

  int buf = 0;
  for (int kb = 0; kb < K; kb += 32) {
    const bool more = (kb + 32) < K;
    if (more) {
      if constexpr (A16P) a_async(kb + 32, buf ^ 1);
      else                a_load(kb + 32);
      b_load(kb + 32);
    }

    v16h bf[2];
    #pragma unroll
    for (int ni = 0; ni < 2; ni++)
      bf[ni] = ld_frag(&Blds[buf][(waveN + ni * 16 + ln) * 32 + lg * 8], 16);
    #pragma unroll
    for (int mi = 0; mi < 4; mi++) {
      v16h af = ld_frag(&Alds[buf][(waveM + mi * 16 + ln) * 32 + lg * 8], 16);
      #pragma unroll
      for (int ni = 0; ni < 2; ni++)
        acc[mi][ni] = wmma_f16(af, bf[ni], acc[mi][ni]);
    }

    if (more) {
      if constexpr (!A16P) a_store(buf ^ 1);
      b_store(buf ^ 1);
      if constexpr (A16P) asm volatile("s_wait_asynccnt 0x0" ::: "memory");
    }
    __syncthreads();
    buf ^= 1;
  }

  // epilogue: C layout VGPR r -> row r + lg*8, lane -> col
  #pragma unroll
  for (int mi = 0; mi < 4; mi++) {
    #pragma unroll
    for (int ni = 0; ni < 2; ni++) {
      const int col = blockN + waveN + ni * 16 + ln;
      const float bv = bias ? bias[col] : 0.0f;
      if constexpr (TRANS) {
        // rows r=0..7 are consecutive -> one packed b128 store per tile
        v8h pk;
        #pragma unroll
        for (int r = 0; r < 8; r++) pk[r] = (_Float16)(acc[mi][ni][r] + bv);
        const int row0 = blockM + waveM + mi * 16 + lg * 8;
        *(v8h*)&out16[(size_t)col * M + row0] = pk;
      } else {
        #pragma unroll
        for (int r = 0; r < 8; r++) {
          const int row = blockM + waveM + mi * 16 + r + lg * 8;
          const float v = acc[mi][ni][r] + bv;
          if constexpr (OUT16) out16[(size_t)row * N + col] = (_Float16)v;
          else                 out32[(size_t)row * N + col] = v;
        }
      }
    }
  }
}

// ---------------------------------------------------------------------------
// RoPE (interleaved pairs), in-place on f16 [B*S, heads, 128]
// ---------------------------------------------------------------------------
__global__ __launch_bounds__(256)
void rope_kernel(_Float16* __restrict__ data, const float* __restrict__ cosT,
                 const float* __restrict__ sinT, const int* __restrict__ start_pos,
                 int S, int heads, int total) {
  const int idx = blockIdx.x * blockDim.x + threadIdx.x;
  if (idx >= total) return;
  const int i    = idx & 63;
  const int h    = (idx >> 6) % heads;
  const int srow = (idx >> 6) / heads;
  const int s    = srow % S;
  const int sp   = *start_pos;
  const float c  = cosT[(size_t)(sp + s) * 64 + i];
  const float sn = sinT[(size_t)(sp + s) * 64 + i];
  const size_t base = ((size_t)srow * heads + h) * 128 + (size_t)i * 2;
  const float x0 = (float)data[base];
  const float x1 = (float)data[base + 1];
  data[base]     = (_Float16)(x0 * c - x1 * sn);
  data[base + 1] = (_Float16)(x0 * sn + x1 * c);
}

// ---------------------------------------------------------------------------
// Flash attention (causal, GQA group=4), f16 in / f16 out.
// 4 waves/block, one 16-row q tile per wave, 32-key chunks.
// q: [B*S, 32, 128] (A-frag pattern direct from global)
// k: [B*S, 8, 128]  (B-frag pattern direct from global, contiguous along hd)
// v: [B, 8, 128, S] TRANSPOSED (B-frag pattern direct from global, contiguous
//    along keys) -> no LDS staging / register transpose in the hot loop.
// Only P does an LDS round-trip (C-layout -> A-layout).
// ---------------------------------------------------------------------------
__global__ __launch_bounds__(128)
void flash_attn_kernel(const _Float16* __restrict__ q16,
                       const _Float16* __restrict__ k16,
                       const _Float16* __restrict__ vt16,
                       _Float16* __restrict__ o16, int S) {
  __shared__ _Float16 PAll[4 * 16 * 32];    // per-wave P tile: [m][key32]

  const int tid  = threadIdx.x;
  const int lane = tid & 31;
  const int wave = tid >> 5;
  const int lg   = lane >> 4;
  const int ln   = lane & 15;
  _Float16* P = PAll + wave * (16 * 32);

  const int b  = blockIdx.z;
  const int hq = blockIdx.y;
  const int hk = hq >> 2;
  const int qt = blockIdx.x * 4 + wave;
  const int q0 = qt * 16;

  // Per-head transposed V base: vt16[b][hk][d][s]
  const _Float16* vtb = vt16 + ((size_t)b * 8 + hk) * 128 * (size_t)S;

  // Preload Q fragments
  v16h qf[4];
  {
    const _Float16* qrow = q16 + (((size_t)b * S + q0 + ln) * 32 + hq) * 128;
    #pragma unroll
    for (int hc = 0; hc < 4; hc++)
      qf[hc] = ld_frag(qrow + hc * 32 + lg * 8, 16);
  }

  v8f o_acc[8];
  #pragma unroll
  for (int nt = 0; nt < 8; nt++) o_acc[nt] = v8f{};
  float mrow[8], lrow[8];
  #pragma unroll
  for (int r = 0; r < 8; r++) { mrow[r] = -3.0e38f; lrow[r] = 0.0f; }

  const float scale = 0.08838834764831845f;   // 1/sqrt(128)
  const int nkc = ((q0 + 15) >> 5) + 1;       // causal chunk count

  #pragma clang loop unroll(disable)
  for (int kc = 0; kc < nkc; kc++) {
    const int kbase = kc * 32;

    // ---- S = Q K^T for two 16-key tiles (K direct from global) ----
    v8f sacc[2];
    sacc[0] = v8f{}; sacc[1] = v8f{};
    #pragma unroll
    for (int t = 0; t < 2; t++) {
      const _Float16* krow = k16 + (((size_t)b * S + kbase + t * 16 + ln) * 8 + hk) * 128;
      #pragma unroll
      for (int hc = 0; hc < 4; hc++) {
        v16h bf = ld_frag(krow + hc * 32 + lg * 8, 16);
        sacc[t] = wmma_f16(qf[hc], bf, sacc[t]);
      }
    }

    // ---- online softmax (row stats across 16-lane groups) ----
    float pv[2][8], alpha[8];
    #pragma unroll
    for (int r = 0; r < 8; r++) {
      const int qi = q0 + r + lg * 8;
      float rm = -3.0e38f;
      #pragma unroll
      for (int t = 0; t < 2; t++) {
        const int kj = kbase + t * 16 + ln;
        float sv = sacc[t][r] * scale;
        if (kj > qi) sv = -3.0e38f;
        sacc[t][r] = sv;
        rm = fmaxf(rm, sv);
      }
      rm = fmaxf(rm, __shfl_xor(rm, 1, 32));
      rm = fmaxf(rm, __shfl_xor(rm, 2, 32));
      rm = fmaxf(rm, __shfl_xor(rm, 4, 32));
      rm = fmaxf(rm, __shfl_xor(rm, 8, 32));
      const float mo = mrow[r];
      const float mn = fmaxf(mo, rm);
      const float al = __expf(mo - mn);
      alpha[r] = al;
      float rs = 0.0f;
      #pragma unroll
      for (int t = 0; t < 2; t++) {
        const float p = __expf(sacc[t][r] - mn);
        pv[t][r] = p;
        rs += p;
      }
      rs += __shfl_xor(rs, 1, 32);
      rs += __shfl_xor(rs, 2, 32);
      rs += __shfl_xor(rs, 4, 32);
      rs += __shfl_xor(rs, 8, 32);
      lrow[r] = lrow[r] * al + rs;
      mrow[r] = mn;
    }

    #pragma unroll
    for (int nt = 0; nt < 8; nt++)
      #pragma unroll
      for (int r = 0; r < 8; r++) o_acc[nt][r] *= alpha[r];

    // ---- P: C layout -> LDS -> A-fragment layout ----
    #pragma unroll
    for (int t = 0; t < 2; t++)
      #pragma unroll
      for (int r = 0; r < 8; r++)
        P[(r + lg * 8) * 32 + t * 16 + ln] = (_Float16)pv[t][r];

    v16h pf = ld_frag(P + ln * 32 + lg * 8, 16);

    // ---- O += P @ V  (V^T direct from global: contiguous along keys) ----
    #pragma unroll
    for (int nt = 0; nt < 8; nt++) {
      v16h vf = ld_frag(vtb + (size_t)(nt * 16 + ln) * S + kbase + lg * 8, 16);
      o_acc[nt] = wmma_f16(pf, vf, o_acc[nt]);
    }
  }

  // ---- normalize & store [b, s, 32, 128] f16 ----
  #pragma unroll
  for (int nt = 0; nt < 8; nt++) {
    #pragma unroll
    for (int r = 0; r < 8; r++) {
      const int qi = q0 + r + lg * 8;
      const size_t off = (((size_t)b * S + qi) * 32 + hq) * 128 + nt * 16 + ln;
      o16[off] = (_Float16)(o_acc[nt][r] / lrow[r]);
    }
  }
}

// ---------------------------------------------------------------------------
// Launch
// ---------------------------------------------------------------------------
extern "C" void kernel_launch(void* const* d_in, const int* in_sizes, int n_in,
                              void* d_out, int out_size, void* d_ws, size_t ws_size,
                              hipStream_t stream) {
  const float* x  = (const float*)d_in[0];
  const float* fc = (const float*)d_in[1];
  const float* fs = (const float*)d_in[2];
  const float* qw = (const float*)d_in[3];
  const float* qb = (const float*)d_in[4];
  const float* kw = (const float*)d_in[5];
  const float* kb = (const float*)d_in[6];
  const float* vw = (const float*)d_in[7];
  const float* vb = (const float*)d_in[8];
  const float* ow = (const float*)d_in[9];
  const int* start_pos = (const int*)d_in[10];
  float* out = (float*)d_out;

  const int D  = 4096;
  const int HD = 128;
  const int S  = in_sizes[1] / (HD / 2);
  const int Mr = in_sizes[0] / D;          // B*S
  const int B  = Mr / S;
  const int NKV = 8 * HD;                  // 1024

  _Float16* q16  = (_Float16*)d_ws;                 // [B*S, 32, 128]
  _Float16* k16  = q16  + (size_t)Mr * D;           // [B*S, 8, 128]
  _Float16* vt16 = k16  + (size_t)Mr * NKV;         // [B, 8, 128, S] transposed
  _Float16* a16  = vt16 + (size_t)Mr * NKV;         // [B*S, 32, 128]

  const dim3 blk(256);
  // Q/K projections (f16 out, fused bias, row-major)
  wmma_gemm_kernel<false, true, false><<<dim3(D   / 128, Mr / 128), blk, 0, stream>>>(x, nullptr, qw, qb, q16, nullptr, Mr, D,   D);
  wmma_gemm_kernel<false, true, false><<<dim3(NKV / 128, Mr / 128), blk, 0, stream>>>(x, nullptr, kw, kb, k16, nullptr, Mr, NKV, D);
  // V projection, per batch, stored TRANSPOSED per head: vt16[b][hk][d][s]
  for (int b = 0; b < B; b++)
    wmma_gemm_kernel<false, true, true><<<dim3(NKV / 128, S / 128), blk, 0, stream>>>(
        x + (size_t)b * S * D, nullptr, vw, vb, vt16 + (size_t)b * NKV * S, nullptr, S, NKV, D);

  const int totq = Mr * 32 * 64;
  const int totk = Mr * 8 * 64;
  rope_kernel<<<(totq + 255) / 256, 256, 0, stream>>>(q16, fc, fs, start_pos, S, 32, totq);
  rope_kernel<<<(totk + 255) / 256, 256, 0, stream>>>(k16, fc, fs, start_pos, S, 8,  totk);

  flash_attn_kernel<<<dim3(S / 64, 32, B), dim3(128), 0, stream>>>(q16, k16, vt16, a16, S);

  // Output projection: f16 A staged via global_load_async_to_lds_b128
  wmma_gemm_kernel<true, false, false><<<dim3(D / 128, Mr / 128), blk, 0, stream>>>(nullptr, a16, ow, nullptr, nullptr, out, Mr, D, D);
}
